// CrossAxialAttention_37890201485525
// MI455X (gfx1250) — compile-verified, matrix operand order checked by reading, not run on hardware
//
#include <hip/hip_runtime.h>

typedef unsigned short u16;
typedef __attribute__((ext_vector_type(16))) __bf16 v16bf;
typedef __attribute__((ext_vector_type(8)))  __bf16 v8bf;
typedef __attribute__((ext_vector_type(8)))  float  v8f;

#define DEV __device__ __forceinline__

// ---------------- problem constants ----------------
// B=16, D1=D2=128, H=W=96 (post-pool), H2F=W2F=192
static constexpr size_t NPC = (size_t)16 * 128 * 96 * 96;   // 18,874,368 elems
static constexpr size_t OUT_N  = (size_t)16 * 128 * 192 * 192; // 75,497,472
static constexpr size_t A1_OFF = OUT_N;                     // a1: 1536*128*128
static constexpr size_t A2_OFF = A1_OFF + (size_t)1536*128*128;
static constexpr size_t XF_OFF = A2_OFF + (size_t)16*128*128;

// ---------------- workspace byte offsets ----------------
static constexpr size_t WB = 16384 * 2;                     // one 128x128 bf16 weight
static constexpr size_t OFF_WQH = 0;
static constexpr size_t OFF_WKH = OFF_WQH + WB;
static constexpr size_t OFF_WVH = OFF_WKH + WB;
static constexpr size_t OFF_WQW = OFF_WVH + WB;
static constexpr size_t OFF_WKW = OFF_WQW + WB;
static constexpr size_t OFF_WVW = OFF_WKW + WB;
static constexpr size_t OFF_WC  = OFF_WVW + WB;
static constexpr size_t OFF_CHB = OFF_WC + WB;              // ch bf16 [B,W,H,D]
static constexpr size_t OFF_XHB = OFF_CHB + NPC * 2;        // xh bf16 [B,W,H,D]
static constexpr size_t OFF_XWF = OFF_XHB + NPC * 2;        // xw f32  [B,D,H,W]
static constexpr size_t OFF_XWB = OFF_XWF + NPC * 4;        // xw bf16 [B,HW,D]
static constexpr size_t OFF_CWB = OFF_XWB + NPC * 2;        // cw bf16 [B,HW,D]
static constexpr size_t OFF_Q2  = OFF_CWB + NPC * 2;        // q2 bf16 [B,128,HW]
static constexpr size_t OFF_K2  = OFF_Q2 + NPC * 2;         // k2 bf16 [B,128,HW]
static constexpr size_t OFF_V2  = OFF_K2 + NPC * 2;         // v2t bf16 [B,HW,128]
static constexpr size_t OFF_P2  = OFF_V2 + NPC * 2;         // P2 bf16 [B,128,128]
static constexpr size_t OFF_O2  = OFF_P2 + (size_t)16*128*128*2; // o2 bf16 [B,128,HW]
static constexpr size_t OFF_ZF  = OFF_O2 + NPC * 2;         // z f32 [B,D,96,96]

// ---------------- helpers ----------------
DEV u16 f2bfu(float f){
  unsigned u = __float_as_uint(f);
  u += 0x7FFFu + ((u >> 16) & 1u);            // round-to-nearest-even
  return (u16)(u >> 16);
}
DEV v8f vzero(){ v8f z = {0.f,0.f,0.f,0.f,0.f,0.f,0.f,0.f}; return z; }

DEV v8f wmma_bf16(v16bf a, v16bf b, v8f c){
  return __builtin_amdgcn_wmma_f32_16x16x32_bf16(false, a, false, b, (short)0, c, false, false);
}

// Fragment with table-order k per lane (k=8h+s, 16+8h+s-8) (A: val(m,k)=src[m*ld+k];
// B via transposed source: val(k,n)=src[n*ld+k]).  Two b128 loads per fragment.
// Requires 16-byte alignment of src rows (ld multiple of 8 u16, k0 multiple of 8).
DEV v16bf ldfrag_ck(const u16* src, size_t ld, int row0, int k0){
  int lane = threadIdx.x & 31;
  int row = row0 + (lane & 15), half = lane >> 4;
  const u16* p = src + (size_t)row * ld + k0 + 8 * half;
  uint4 lo = *(const uint4*)p;
  uint4 hi = *(const uint4*)(p + 16);
  v8bf l8 = __builtin_bit_cast(v8bf, lo);
  v8bf h8 = __builtin_bit_cast(v8bf, hi);
  return __builtin_shufflevector(l8, h8, 0,1,2,3,4,5,6,7,8,9,10,11,12,13,14,15);
}

// Pack a D-fragment column (8 f32 along M) into 8 bf16 = one uint4
DEV uint4 pack8bf(const v8f a){
  uint4 o;
  o.x = (unsigned)f2bfu(a[0]) | ((unsigned)f2bfu(a[1]) << 16);
  o.y = (unsigned)f2bfu(a[2]) | ((unsigned)f2bfu(a[3]) << 16);
  o.z = (unsigned)f2bfu(a[4]) | ((unsigned)f2bfu(a[5]) << 16);
  o.w = (unsigned)f2bfu(a[6]) | ((unsigned)f2bfu(a[7]) << 16);
  return o;
}

// Positional encodings (d_model=128, dm=64). ln(10000)/64
DEV float peH(int d, int h){
  if (d < 64) return (d & 1) ? 1.0f : 0.0f;        // width term at w=0
  int j = (d - 64) >> 1;
  float dv = __expf(-(float)(2 * j) * (9.210340371976184f / 64.0f));
  return (d & 1) ? __cosf((float)h * dv) : __sinf((float)h * dv);
}
DEV float peW(int d, int w){
  if (d >= 64) return (d & 1) ? 1.0f : 0.0f;       // height term at h=0
  int j = d >> 1;
  float dv = __expf(-(float)(2 * j) * (9.210340371976184f / 64.0f));
  return (d & 1) ? __cosf((float)w * dv) : __sinf((float)w * dv);
}

// ---------------- kernels ----------------
__global__ void k_cvt_w(const float* __restrict__ src, u16* __restrict__ dst, int n){
  int i = blockIdx.x * 256 + threadIdx.x;
  if (i < n) dst[i] = f2bfu(src[i]);
}

// maxpool(2,2)+PE. idx over [b,d,h,w] natural (coalesced xwf writes).
// xhb: [B,W,H,D] (k-contiguous for stage-1 B operands)
// xwf: [B,D,H,W] f32 (for xf / final gather)
// xwb: [B,HW,D] bf16 (k-contiguous for stage-2 B operands)
__global__ void k_prep_x(const float* __restrict__ x, u16* __restrict__ xhb,
                         float* __restrict__ xwf, u16* __restrict__ xwb){
  size_t idx = (size_t)blockIdx.x * 256 + threadIdx.x;
  if (idx >= NPC) return;
  int w = (int)(idx % 96); int h = (int)((idx / 96) % 96);
  int d = (int)((idx / 9216) % 128); int b = (int)(idx / 1179648);
  const float* p = x + (((size_t)b * 128 + d) * 192 + 2 * h) * 192 + 2 * w;
  float m = fmaxf(fmaxf(p[0], p[1]), fmaxf(p[192], p[193]));
  float vh = m + peH(d, h);
  float vw = vh + peW(d, w);
  xhb[(((size_t)b * 96 + w) * 96 + h) * 128 + d] = f2bfu(vh);
  xwf[idx] = vw;
  xwb[((size_t)b * 9216 + h * 96 + w) * 128 + d] = f2bfu(vw);
}

// nearest-upsample 48->96 + height-PE -> ch bf16 [B,W,H,D]
// idx over [b,w,h,d] so stores are coalesced.
__global__ void k_prep_c(const float* __restrict__ ctx, u16* __restrict__ chb){
  size_t idx = (size_t)blockIdx.x * 256 + threadIdx.x;
  if (idx >= NPC) return;
  int d = (int)(idx % 128); int h = (int)((idx / 128) % 96);
  int w = (int)((idx / 12288) % 96); int b = (int)(idx / 1179648);
  float v = ctx[(((size_t)b * 128 + d) * 48 + (h >> 1)) * 48 + (w >> 1)] + peH(d, h);
  chb[idx] = f2bfu(v);
}

// Stage 1: per (b,w) column, 8 waves. All operands consumed k-contiguous.
__global__ __launch_bounds__(256) void k_stage1(
    const u16* __restrict__ chb, const u16* __restrict__ xhb,
    const u16* __restrict__ wqb, const u16* __restrict__ wkb, const u16* __restrict__ wvb,
    float* __restrict__ a1_out, u16* __restrict__ cwb){
  extern __shared__ char smem[];
  u16* buf0 = (u16*)smem;               // ch_t [96 h][128 d] -> Q [128 i][96 h]
  u16* buf1 = (u16*)(smem + 24576);     // K [128 j][96 h]
  u16* buf2 = (u16*)(smem + 49152);     // xh_t [96 h][128 d] -> V_t [96 h][128 j]
  u16* bufP = (u16*)smem;               // P [128 i][128 j] (overwrites buf0+1/3 of buf1)
  int tid = threadIdx.x, wave = tid >> 5, lane = tid & 31, nn = lane & 15, half = lane >> 4;
  int bw = blockIdx.x, b = bw / 96, w = bw % 96, mt = wave;

  { // cooperative 128-bit tile loads
    const uint4* chg4 = (const uint4*)(chb + (size_t)bw * 12288);
    const uint4* xhg4 = (const uint4*)(xhb + (size_t)bw * 12288);
    uint4* b04 = (uint4*)buf0; uint4* b24 = (uint4*)buf2;
    for (int t = tid; t < 1536; t += 256){ b04[t] = chg4[t]; b24[t] = xhg4[t]; }
  }
  __syncthreads();

  { // Q = Wq_h * ch: A=weights [m][k], B=ch_t [h][d] (val(k=d,n=h) k-contig)
    v8f acc[6];
#pragma unroll
    for (int nt = 0; nt < 6; ++nt) acc[nt] = vzero();
    for (int ks = 0; ks < 4; ++ks){
      v16bf a = ldfrag_ck(wqb, 128, mt * 16, ks * 32);
#pragma unroll
      for (int nt = 0; nt < 6; ++nt)
        acc[nt] = wmma_bf16(a, ldfrag_ck(buf0, 128, nt * 16, ks * 32), acc[nt]);
    }
    __syncthreads();                     // everyone done reading ch_t
#pragma unroll
    for (int nt = 0; nt < 6; ++nt)       // Q stored [i][h] (k-contig for scores)
#pragma unroll
      for (int r = 0; r < 8; ++r)
        buf0[(mt * 16 + r + 8 * half) * 96 + nt * 16 + nn] = f2bfu(acc[nt][r]);
  }
  { // K = Wk_h * xh -> buf1 [j][h]
    v8f acc[6];
#pragma unroll
    for (int nt = 0; nt < 6; ++nt) acc[nt] = vzero();
    for (int ks = 0; ks < 4; ++ks){
      v16bf a = ldfrag_ck(wkb, 128, mt * 16, ks * 32);
#pragma unroll
      for (int nt = 0; nt < 6; ++nt)
        acc[nt] = wmma_bf16(a, ldfrag_ck(buf2, 128, nt * 16, ks * 32), acc[nt]);
    }
#pragma unroll
    for (int nt = 0; nt < 6; ++nt)
#pragma unroll
      for (int r = 0; r < 8; ++r)
        buf1[(mt * 16 + r + 8 * half) * 96 + nt * 16 + nn] = f2bfu(acc[nt][r]);
  }
  { // V = Wv_h * xh, stored TRANSPOSED V_t [h][j] (packed uint4 stores)
    v8f acc[6];
#pragma unroll
    for (int nt = 0; nt < 6; ++nt) acc[nt] = vzero();
    for (int ks = 0; ks < 4; ++ks){
      v16bf a = ldfrag_ck(wvb, 128, mt * 16, ks * 32);
#pragma unroll
      for (int nt = 0; nt < 6; ++nt)
        acc[nt] = wmma_bf16(a, ldfrag_ck(buf2, 128, nt * 16, ks * 32), acc[nt]);
    }
    __syncthreads();                     // all reads of xh_t done
#pragma unroll
    for (int nt = 0; nt < 6; ++nt)
      *(uint4*)&buf2[(nt * 16 + nn) * 128 + mt * 16 + 8 * half] = pack8bf(acc[nt]);
    __syncthreads();                     // Q,K,V visible
  }
  // S = Q*K^T (K-dim=96): A=Q[i][h] k-contig, B=K[j][h] k-contig
  {
    v8f accS[8];
#pragma unroll
    for (int nt = 0; nt < 8; ++nt) accS[nt] = vzero();
    for (int ks = 0; ks < 3; ++ks){
      v16bf a = ldfrag_ck(buf0, 96, mt * 16, ks * 32);
#pragma unroll
      for (int nt = 0; nt < 8; ++nt)
        accS[nt] = wmma_bf16(a, ldfrag_ck(buf1, 96, nt * 16, ks * 32), accS[nt]);
    }
    float* a1b = a1_out + (size_t)bw * 16384;
#pragma unroll
    for (int r = 0; r < 8; ++r){
      float mx = -1e30f;
#pragma unroll
      for (int nt = 0; nt < 8; ++nt) mx = fmaxf(mx, accS[nt][r]);
#pragma unroll
      for (int s = 1; s < 16; s <<= 1) mx = fmaxf(mx, __shfl_xor(mx, s, 32));
      float sum = 0.f;
#pragma unroll
      for (int nt = 0; nt < 8; ++nt){ float e = __expf(accS[nt][r] - mx); accS[nt][r] = e; sum += e; }
#pragma unroll
      for (int s = 1; s < 16; s <<= 1) sum += __shfl_xor(sum, s, 32);
      float inv = 1.0f / sum;
      int m = mt * 16 + r + 8 * half;
#pragma unroll
      for (int nt = 0; nt < 8; ++nt){
        float p = accS[nt][r] * inv; accS[nt][r] = p;
        a1b[(size_t)m * 128 + nt * 16 + nn] = p;
      }
    }
    __syncthreads();                     // all reads of Q/K done before P overwrites
#pragma unroll
    for (int nt = 0; nt < 8; ++nt)       // P [i][j] (k-contig for O GEMM)
#pragma unroll
      for (int r = 0; r < 8; ++r)
        bufP[(mt * 16 + r + 8 * half) * 128 + nt * 16 + nn] = f2bfu(accS[nt][r]);
    __syncthreads();
  }
  // O = P*V: A=P[i][j] k-contig, B=V_t[h][j] k-contig.  cw = O + peW -> [B,HW,D]
  {
    v8f acc[6];
#pragma unroll
    for (int nt = 0; nt < 6; ++nt) acc[nt] = vzero();
    for (int ks = 0; ks < 4; ++ks){
      v16bf a = ldfrag_ck(bufP, 128, mt * 16, ks * 32);
#pragma unroll
      for (int nt = 0; nt < 6; ++nt)
        acc[nt] = wmma_bf16(a, ldfrag_ck(buf2, 128, nt * 16, ks * 32), acc[nt]);
    }
#pragma unroll
    for (int nt = 0; nt < 6; ++nt){
      int h = nt * 16 + nn;
      v8f o;
#pragma unroll
      for (int r = 0; r < 8; ++r) o[r] = acc[nt][r] + peW(mt * 16 + r + 8 * half, w);
      *(uint4*)&cwb[((size_t)b * 9216 + h * 96 + w) * 128 + mt * 16 + 8 * half] = pack8bf(o);
    }
  }
}

// Stage 2 QKV: per (b, 128-col chunk of HW). Pure uint4 LDS staging.
__global__ __launch_bounds__(256) void k_stage2_qkv(
    const u16* __restrict__ cwb, const u16* __restrict__ xwb,
    const u16* __restrict__ wqb, const u16* __restrict__ wkb, const u16* __restrict__ wvb,
    u16* __restrict__ q2, u16* __restrict__ k2, u16* __restrict__ v2t){
  extern __shared__ char smem[];
  u16* sC = (u16*)smem;                 // cw_t [128 n][128 d]
  u16* sX = (u16*)(smem + 32768);       // xw_t [128 n][128 d]
  int b = blockIdx.y, n0 = blockIdx.x * 128;
  int tid = threadIdx.x, wave = tid >> 5, lane = tid & 31, nn = lane & 15, half = lane >> 4;
  {
    const uint4* c4 = (const uint4*)(cwb + ((size_t)b * 9216 + n0) * 128);
    const uint4* x4 = (const uint4*)(xwb + ((size_t)b * 9216 + n0) * 128);
    uint4* sc4 = (uint4*)sC; uint4* sx4 = (uint4*)sX;
    for (int t = tid; t < 2048; t += 256){ sc4[t] = c4[t]; sx4[t] = x4[t]; }
  }
  __syncthreads();
  int mt = wave;
  const u16* Ws[3] = {wqb, wkb, wvb};
  const u16* Bs[3] = {sC, sX, sX};
  for (int g = 0; g < 3; ++g){
    v8f acc[8];
#pragma unroll
    for (int nt = 0; nt < 8; ++nt) acc[nt] = vzero();
    for (int ks = 0; ks < 4; ++ks){
      v16bf a = ldfrag_ck(Ws[g], 128, mt * 16, ks * 32);
#pragma unroll
      for (int nt = 0; nt < 8; ++nt)
        acc[nt] = wmma_bf16(a, ldfrag_ck(Bs[g], 128, nt * 16, ks * 32), acc[nt]);
    }
    if (g == 2){                         // v stored transposed [B,HW,128]: packed
#pragma unroll
      for (int nt = 0; nt < 8; ++nt){
        int n = n0 + nt * 16 + nn;
        *(uint4*)&v2t[((size_t)b * 9216 + n) * 128 + mt * 16 + 8 * half] = pack8bf(acc[nt]);
      }
    } else {
      u16* O = (g == 0) ? q2 : k2;       // [B,128,HW] (k-contig rows for scores)
#pragma unroll
      for (int nt = 0; nt < 8; ++nt)
#pragma unroll
        for (int r = 0; r < 8; ++r){
          int m = mt * 16 + r + 8 * half;
          O[((size_t)b * 128 + m) * 9216 + n0 + nt * 16 + nn] = f2bfu(acc[nt][r]);
        }
    }
  }
}

// Stage 2 scores: S2 = q2*k2^T over K=9216 (rows stream from L2 as b128 loads)
__global__ __launch_bounds__(256) void k_stage2_scores(
    const u16* __restrict__ q2, const u16* __restrict__ k2,
    float* __restrict__ a2_out, u16* __restrict__ p2){
  int b = blockIdx.x;
  int tid = threadIdx.x, wave = tid >> 5, lane = tid & 31, nn = lane & 15, half = lane >> 4;
  int mt = wave;
  const u16* qb = q2 + (size_t)b * 128 * 9216;
  const u16* kb = k2 + (size_t)b * 128 * 9216;
  v8f acc[8];
#pragma unroll
  for (int nt = 0; nt < 8; ++nt) acc[nt] = vzero();
  for (int ks = 0; ks < 288; ++ks){
    if (ks + 8 < 288)
      __builtin_prefetch(qb + (size_t)(mt * 16 + nn) * 9216 + (ks + 8) * 32, 0, 0);
    v16bf a = ldfrag_ck(qb, 9216, mt * 16, ks * 32);
#pragma unroll
    for (int nt = 0; nt < 8; ++nt)
      acc[nt] = wmma_bf16(a, ldfrag_ck(kb, 9216, nt * 16, ks * 32), acc[nt]);
  }
  float* a2b = a2_out + (size_t)b * 16384;
  u16* p2b = p2 + (size_t)b * 16384;
#pragma unroll
  for (int r = 0; r < 8; ++r){
    float mx = -1e30f;
#pragma unroll
    for (int nt = 0; nt < 8; ++nt) mx = fmaxf(mx, acc[nt][r]);
#pragma unroll
    for (int s = 1; s < 16; s <<= 1) mx = fmaxf(mx, __shfl_xor(mx, s, 32));
    float sum = 0.f;
#pragma unroll
    for (int nt = 0; nt < 8; ++nt){ float e = __expf(acc[nt][r] - mx); acc[nt][r] = e; sum += e; }
#pragma unroll
    for (int s = 1; s < 16; s <<= 1) sum += __shfl_xor(sum, s, 32);
    float inv = 1.0f / sum;
    int m = mt * 16 + r + 8 * half;
#pragma unroll
    for (int nt = 0; nt < 8; ++nt){
      float p = acc[nt][r] * inv;
      a2b[(size_t)m * 128 + nt * 16 + nn] = p;
      p2b[(size_t)m * 128 + nt * 16 + nn] = f2bfu(p);
    }
  }
}

// Stage 2 AV: o2 = P2 * v2, v2t chunk staged via uint4, B k-contiguous
__global__ __launch_bounds__(256) void k_stage2_av(
    const u16* __restrict__ p2, const u16* __restrict__ v2t, u16* __restrict__ o2){
  extern __shared__ char smem[];
  u16* sV = (u16*)smem;                 // [128 n][128 j]
  int b = blockIdx.y, n0 = blockIdx.x * 128;
  int tid = threadIdx.x, wave = tid >> 5, lane = tid & 31, nn = lane & 15, half = lane >> 4;
  {
    const uint4* v4 = (const uint4*)(v2t + ((size_t)b * 9216 + n0) * 128);
    uint4* s4 = (uint4*)sV;
    for (int t = tid; t < 2048; t += 256) s4[t] = v4[t];
  }
  __syncthreads();
  int mt = wave;
  const u16* pb = p2 + (size_t)b * 16384;
  v8f acc[8];
#pragma unroll
  for (int nt = 0; nt < 8; ++nt) acc[nt] = vzero();
  for (int ks = 0; ks < 4; ++ks){
    v16bf a = ldfrag_ck(pb, 128, mt * 16, ks * 32);
#pragma unroll
    for (int nt = 0; nt < 8; ++nt)
      acc[nt] = wmma_bf16(a, ldfrag_ck(sV, 128, nt * 16, ks * 32), acc[nt]);
  }
#pragma unroll
  for (int nt = 0; nt < 8; ++nt)
#pragma unroll
    for (int r = 0; r < 8; ++r){
      int m = mt * 16 + r + 8 * half;
      o2[((size_t)b * 128 + m) * 9216 + n0 + nt * 16 + nn] = f2bfu(acc[nt][r]);
    }
}

// y = Wc * o_scrambled + bias -> BN -> sigmoid -> * xf  (scramble = flat gather)
__global__ __launch_bounds__(256) void k_fuse_conv(
    const u16* __restrict__ o2, const u16* __restrict__ wcb,
    const float* __restrict__ xwf, const float* __restrict__ bc,
    const float* __restrict__ gam, const float* __restrict__ bet,
    float* __restrict__ zf){
  extern __shared__ char smem[];
  u16* sO = (u16*)smem;                 // [128 n][128 j] (k-contiguous B)
  int b = blockIdx.y, n0 = blockIdx.x * 128;
  int tid = threadIdx.x, wave = tid >> 5, lane = tid & 31, nn = lane & 15, half = lane >> 4;
  const u16* ob = o2 + (size_t)b * 1179648;
  for (int t = tid; t < 16384; t += 256){
    int nl = t >> 7, j = t & 127;
    int n = n0 + nl, i = n / 96, w = n % 96;
    sO[t] = ob[(size_t)i * 12288 + j * 96 + w];   // torch reshape+permute gather
  }
  __syncthreads();
  int mt = wave;
  v8f acc[8];
#pragma unroll
  for (int nt = 0; nt < 8; ++nt) acc[nt] = vzero();
  for (int ks = 0; ks < 4; ++ks){
    v16bf a = ldfrag_ck(wcb, 128, mt * 16, ks * 32);
#pragma unroll
    for (int nt = 0; nt < 8; ++nt)
      acc[nt] = wmma_bf16(a, ldfrag_ck(sO, 128, nt * 16, ks * 32), acc[nt]);
  }
  const float* xb = xwf + (size_t)b * 1179648;
  const float bnr = 0.9999950000374997f;          // 1/sqrt(1 + 1e-5)
#pragma unroll
  for (int nt = 0; nt < 8; ++nt)
#pragma unroll
    for (int r = 0; r < 8; ++r){
      int c = mt * 16 + r + 8 * half;
      int n = n0 + nt * 16 + nn, i = n / 96, w = n % 96;
      float y = acc[nt][r] + bc[c];
      y = y * (gam[c] * bnr) + bet[c];
      float s = 1.0f / (1.0f + __expf(-y));
      float z = s * xb[(size_t)i * 12288 + c * 96 + w];   // * xf (same gather)
      zf[(((size_t)b * 128 + c) * 96 + i) * 96 + w] = z;
    }
}

// Bilinear x2 upsample, align_corners=True (96 -> 192)
__global__ void k_upsample(const float* __restrict__ zf, float* __restrict__ outp){
  size_t idx = (size_t)blockIdx.x * 256 + threadIdx.x;
  if (idx >= OUT_N) return;
  int ox = (int)(idx % 192); int oy = (int)((idx / 192) % 192);
  int c = (int)((idx / 36864) % 128); int b = (int)(idx / 4718592);
  const float fs = 95.0f / 191.0f;
  float ys = oy * fs, xs = ox * fs;
  int y0 = (int)ys; if (y0 > 94) y0 = 94; float wy = ys - y0;
  int x0 = (int)xs; if (x0 > 94) x0 = 94; float wx = xs - x0;
  const float* p = zf + (((size_t)b * 128 + c) * 96 + y0) * 96 + x0;
  float v = p[0]  * (1.f - wy) * (1.f - wx) + p[96] * wy * (1.f - wx)
          + p[1]  * (1.f - wy) * wx          + p[97] * wy * wx;
  outp[idx] = v;
}

// xf output: scrambled reinterpretation of xw
__global__ void k_xf(const float* __restrict__ xwf, float* __restrict__ xf_out){
  size_t idx = (size_t)blockIdx.x * 256 + threadIdx.x;
  if (idx >= NPC) return;
  int w = (int)(idx % 96); int i = (int)((idx / 96) % 96);
  int j = (int)((idx / 9216) % 128); int b = (int)(idx / 1179648);
  xf_out[idx] = xwf[(size_t)b * 1179648 + (size_t)i * 12288 + j * 96 + w];
}

// ---------------- launch ----------------
extern "C" void kernel_launch(void* const* d_in, const int* in_sizes, int n_in,
                              void* d_out, int out_size, void* d_ws, size_t ws_size,
                              hipStream_t stream){
  (void)in_sizes; (void)n_in; (void)out_size; (void)ws_size;
  const float* ctx = (const float*)d_in[0];
  const float* x   = (const float*)d_in[1];
  const float* wqh = (const float*)d_in[2];
  const float* wkh = (const float*)d_in[3];
  const float* wvh = (const float*)d_in[4];
  const float* wqw = (const float*)d_in[5];
  const float* wkw = (const float*)d_in[6];
  const float* wvw = (const float*)d_in[7];
  const float* wc  = (const float*)d_in[8];
  const float* bc  = (const float*)d_in[9];
  const float* gam = (const float*)d_in[10];
  const float* bet = (const float*)d_in[11];
  float* outp = (float*)d_out;
  float* a1   = outp + A1_OFF;
  float* a2   = outp + A2_OFF;
  float* xf   = outp + XF_OFF;

  char* ws = (char*)d_ws;
  u16* wqhB = (u16*)(ws + OFF_WQH); u16* wkhB = (u16*)(ws + OFF_WKH);
  u16* wvhB = (u16*)(ws + OFF_WVH); u16* wqwB = (u16*)(ws + OFF_WQW);
  u16* wkwB = (u16*)(ws + OFF_WKW); u16* wvwB = (u16*)(ws + OFF_WVW);
  u16* wcB  = (u16*)(ws + OFF_WC);
  u16* chB  = (u16*)(ws + OFF_CHB); u16* xhB = (u16*)(ws + OFF_XHB);
  float* xwF = (float*)(ws + OFF_XWF);
  u16* xwB  = (u16*)(ws + OFF_XWB);
  u16* cwB  = (u16*)(ws + OFF_CWB);
  u16* q2B  = (u16*)(ws + OFF_Q2);  u16* k2B = (u16*)(ws + OFF_K2);
  u16* v2B  = (u16*)(ws + OFF_V2);  u16* p2B = (u16*)(ws + OFF_P2);
  u16* o2B  = (u16*)(ws + OFF_O2);
  float* zF = (float*)(ws + OFF_ZF);

  k_cvt_w<<<64, 256, 0, stream>>>(wqh, wqhB, 16384);
  k_cvt_w<<<64, 256, 0, stream>>>(wkh, wkhB, 16384);
  k_cvt_w<<<64, 256, 0, stream>>>(wvh, wvhB, 16384);
  k_cvt_w<<<64, 256, 0, stream>>>(wqw, wqwB, 16384);
  k_cvt_w<<<64, 256, 0, stream>>>(wkw, wkwB, 16384);
  k_cvt_w<<<64, 256, 0, stream>>>(wvw, wvwB, 16384);
  k_cvt_w<<<64, 256, 0, stream>>>(wc,  wcB,  16384);

  k_prep_x<<<73728, 256, 0, stream>>>(x, xhB, xwF, xwB);
  k_prep_c<<<73728, 256, 0, stream>>>(ctx, chB);

  k_stage1<<<1536, 256, 73728, stream>>>(chB, xhB, wqhB, wkhB, wvhB, a1, cwB);

  k_stage2_qkv<<<dim3(72, 16), 256, 65536, stream>>>(cwB, xwB, wqwB, wkwB, wvwB,
                                                     q2B, k2B, v2B);
  k_stage2_scores<<<16, 256, 0, stream>>>(q2B, k2B, a2, p2B);
  k_stage2_av<<<dim3(72, 16), 256, 32768, stream>>>(p2B, v2B, o2B);
  k_fuse_conv<<<dim3(72, 16), 256, 32768, stream>>>(o2B, wcB, xwF, bc, gam, bet, zF);

  k_upsample<<<294912, 256, 0, stream>>>(zF, outp);
  k_xf<<<73728, 256, 0, stream>>>(xwF, xf);
}
